// TDNN_65979287601455
// MI455X (gfx1250) — matrix operand order repaired and checked by reading, not compiled, not run on hardware
//
#include <hip/hip_runtime.h>

// ---------------- types ----------------
typedef __attribute__((ext_vector_type(16))) __bf16 v16bf;
typedef __attribute__((ext_vector_type(8)))  float  v8f;
typedef __attribute__((ext_vector_type(4)))  float  f4;

// ---------------- problem constants ----------------
#define B_   32
#define T_   2000
#define DIN  512
#define DOUT 512
#define KW   5
#define TOUT (T_ - KW + 1)       // 1996

#define MT   128                 // M tile per block (8 wmma subtiles)
#define NT   64                  // time tile per block (4 wmma subtiles)
#define NKC  80                  // reduction chunks: KW * (DIN/32)
#define BPITCH 80                // LDS row pitch in bytes (64B data + 16B skew)

// fp32 -> bf16, round-to-nearest-even
__device__ __forceinline__ unsigned short f2bf(float f) {
  union { float f; unsigned u; } v; v.f = f;
  unsigned r = v.u + 0x7FFFu + ((v.u >> 16) & 1u);
  return (unsigned short)(r >> 16);
}

// ---------------- kernel 1: x fp32 -> bf16 ----------------
__global__ __launch_bounds__(256) void cvt_x(const float* __restrict__ x,
                                             unsigned short* __restrict__ xb) {
  size_t i = ((size_t)blockIdx.x * 256 + threadIdx.x) * 8;
  f4 a = *(const f4*)(x + i);
  f4 b = *(const f4*)(x + i + 4);
  union { unsigned short s[8]; f4 v; } o;
  o.s[0] = f2bf(a.x); o.s[1] = f2bf(a.y); o.s[2] = f2bf(a.z); o.s[3] = f2bf(a.w);
  o.s[4] = f2bf(b.x); o.s[5] = f2bf(b.y); o.s[6] = f2bf(b.z); o.s[7] = f2bf(b.w);
  *(f4*)(xb + i) = o.v;
}

// ---------------- kernel 2: weight fp32 [O][I][K] -> bf16 A-fragment layout ----
// Apack index: (((k*16 + ic)*32 + mt)*32 + lane)*16 + j
// ISA 16-bit A 16x32 layout:
//   lanes 0-15 : M=lane,    VGPR v<4: K=2v+{0,1},     v>=4: K=16+2(v-4)+{0,1}
//   lanes 16-31: M=lane-16, VGPR v<4: K=8+2v+{0,1},   v>=4: K=24+2(v-4)+{0,1}
__global__ __launch_bounds__(256) void pack_w(const float* __restrict__ w,
                                              unsigned short* __restrict__ ap) {
  unsigned idx = blockIdx.x * 256 + threadIdx.x;       // < 1,310,720
  unsigned j    =  idx        & 15;
  unsigned lane = (idx >> 4)  & 31;
  unsigned mt   = (idx >> 9)  & 31;
  unsigned ic   = (idx >> 14) & 15;
  unsigned k    =  idx >> 18;                          // 0..4
  unsigned v = j >> 1, h = j & 1;
  unsigned base;
  if (lane < 16) base = (v < 4) ? (2u * v)       : (16u + 2u * (v - 4));
  else           base = (v < 4) ? (8u + 2u * v)  : (24u + 2u * (v - 4));
  unsigned K = base + h;
  unsigned o = mt * 16 + (lane & 15);
  unsigned i = ic * 32 + K;
  ap[idx] = f2bf(w[((size_t)o * DIN + i) * KW + k]);
}

// ---------------- kernel 3: implicit-GEMM conv via bf16 WMMA ----------------
__global__ __launch_bounds__(256) void tdnn_wmma(const unsigned short* __restrict__ xb,
                                                 const unsigned short* __restrict__ ap,
                                                 const float* __restrict__ bias,
                                                 float* __restrict__ out) {
  __shared__ __align__(16) char Bs[NT * BPITCH];

  const int tid  = threadIdx.x;
  const int lane = tid & 31;
  const int wid  = tid >> 5;       // 0..7
  const int wm   = wid & 3;        // wave M group (32 rows)
  const int wn   = wid >> 2;       // wave N group (32 cols)

  const int t0 = blockIdx.x * NT;           // time tile base
  const int mb = blockIdx.y;                // 128-row M block (0..3)
  const int b  = blockIdx.z;                // batch

  const int row = tid >> 2;                 // 0..63  (staging row)
  const int seg = tid & 3;                  // 0..3   (16B segment)

  const int mt0 = mb * 8 + wm * 2;          // first 16-row A subtile owned by wave

  v8f c00 = {}; v8f c01 = {}; v8f c10 = {}; v8f c11 = {};

  const int nrow0 = wn * 32 + (lane & 15);
  const int nrow1 = nrow0 + 16;
  const int hoff  = (lane < 16) ? 0 : 16;   // K half select per B-matrix layout

  for (int kc = 0; kc < NKC; ++kc) {
    const int k  = kc >> 4;                 // conv tap 0..4
    const int ic = kc & 15;                 // D_IN chunk 0..15
    const int i0 = ic * 32;

    __syncthreads();                        // previous chunk's LDS reads done
    {
      int tr = t0 + row + k; if (tr > T_ - 1) tr = T_ - 1;
      f4 v = *(const f4*)(xb + ((size_t)(b * T_ + tr)) * DIN + i0 + seg * 8);
      *(f4*)(Bs + row * BPITCH + seg * 16) = v;
    }
    // prefetch next chunk's x slab into the cache hierarchy
    if (kc + 1 < NKC) {
      int k2 = (kc + 1) >> 4, ic2 = (kc + 1) & 15;
      int tr2 = t0 + row + k2; if (tr2 > T_ - 1) tr2 = T_ - 1;
      __builtin_prefetch(xb + ((size_t)(b * T_ + tr2)) * DIN + ic2 * 32 + seg * 8, 0, 0);
    }
    __syncthreads();                        // slab visible

    // A fragments: per-lane contiguous 32B from the packed (L2-resident) weights
    const unsigned short* aptr =
        ap + ((((size_t)(k * 16 + ic) * 32 + mt0) * 32 + lane) << 4);
    v16bf a0 = *(const v16bf*)aptr;
    v16bf a1 = *(const v16bf*)(aptr + 512);   // next 16-row subtile (32 lanes * 16)

    // B fragments from LDS (conflict-free b128 pairs, 80B skewed pitch)
    union { f4 q[2]; v16bf v; } b0, b1;
    b0.q[0] = *(const f4*)(Bs + nrow0 * BPITCH + hoff);
    b0.q[1] = *(const f4*)(Bs + nrow0 * BPITCH + hoff + 32);
    b1.q[0] = *(const f4*)(Bs + nrow1 * BPITCH + hoff);
    b1.q[1] = *(const f4*)(Bs + nrow1 * BPITCH + hoff + 32);

    c00 = __builtin_amdgcn_wmma_f32_16x16x32_bf16(false, a0, false, b0.v, (short)0, c00, false, false);
    c01 = __builtin_amdgcn_wmma_f32_16x16x32_bf16(false, a0, false, b1.v, (short)0, c01, false, false);
    c10 = __builtin_amdgcn_wmma_f32_16x16x32_bf16(false, a1, false, b0.v, (short)0, c10, false, false);
    c11 = __builtin_amdgcn_wmma_f32_16x16x32_bf16(false, a1, false, b1.v, (short)0, c11, false, false);
  }

  // ---------------- epilogue: bias + predicated store ----------------
  // C/D layout: VGPR r, lanes 0-15 -> M=r, lanes 16-31 -> M=r+8; N = lane&15.
  const int tA = t0 + wn * 32 + (lane & 15);       // n-subtile u=0
  const int tB = tA + 16;                          // n-subtile u=1
  const int mrow = mb * 128 + wm * 32 + ((lane < 16) ? 0 : 8);

#pragma unroll
  for (int r = 0; r < 8; ++r) {
    int o0 = mrow + r;          // s=0 subtile
    int o1 = o0 + 16;           // s=1 subtile
    float bi0 = bias[o0], bi1 = bias[o1];
    if (tA < TOUT) {
      out[((size_t)(b * DOUT + o0)) * TOUT + tA] = c00[r] + bi0;
      out[((size_t)(b * DOUT + o1)) * TOUT + tA] = c10[r] + bi1;
    }
    if (tB < TOUT) {
      out[((size_t)(b * DOUT + o0)) * TOUT + tB] = c01[r] + bi0;
      out[((size_t)(b * DOUT + o1)) * TOUT + tB] = c11[r] + bi1;
    }
  }
}

// ---------------- launcher ----------------
extern "C" void kernel_launch(void* const* d_in, const int* in_sizes, int n_in,
                              void* d_out, int out_size, void* d_ws, size_t ws_size,
                              hipStream_t stream) {
  (void)in_sizes; (void)n_in; (void)out_size; (void)ws_size;
  const float* x    = (const float*)d_in[0];
  const float* w    = (const float*)d_in[1];
  const float* bias = (const float*)d_in[2];
  float* out = (float*)d_out;

  unsigned short* xb = (unsigned short*)d_ws;                    // 32*2000*512 bf16 = 65.5 MB
  unsigned short* ap = xb + (size_t)B_ * T_ * DIN;               // 512*512*5 bf16 = 2.6 MB

  // x: 32,768,000 elems / 8 per thread / 256 per block = 16000 blocks (exact)
  cvt_x<<<16000, 256, 0, stream>>>(x, xb);
  // weights: 1,310,720 elems / 256 = 5120 blocks (exact)
  pack_w<<<5120, 256, 0, stream>>>(w, ap);

  dim3 grid((TOUT + NT - 1) / NT /*32*/, DOUT / MT /*4*/, B_ /*32*/);
  tdnn_wmma<<<grid, 256, 0, stream>>>(xb, ap, bias, out);
}
